// DGI_34291018891273
// MI455X (gfx1250) — matrix operand order repaired and checked by reading, not compiled, not run on hardware
//
#include <hip/hip_runtime.h>

typedef __attribute__((ext_vector_type(16))) _Float16 v16h;
typedef __attribute__((ext_vector_type(8)))  _Float16 v8h;
typedef __attribute__((ext_vector_type(8)))  float    v8f;
typedef __attribute__((ext_vector_type(4)))  float    v4f;

#define N_NODES 8192
#define NIN 128
#define NH  128
#define NC  8
#define CS  1024

__device__ __forceinline__ v8f wmma_f16(v16h a, v16h b, v8f c) {
    // (neg_a, A, neg_b, B, c_mod, C, reuse_a, reuse_b)
    return __builtin_amdgcn_wmma_f32_16x16x32_f16(false, a, false, b, (short)0, c,
                                                  false, false);
}

// Load 16 fp32 from one matrix row (row-major, at column offset k0 already applied)
// and convert to a 16x32 f16 A fragment in contiguous k order (B loads use the same order).
// Lane (r, hi): elements 0..15 -> K = hi*16 + i       (bytes [hi*64,  hi*64+64)  )
// The A-side matrices (adj / seq) are single-use streams: use non-temporal loads
// so the 256MB adj stream does not evict the L2-resident f16 B operands.
__device__ __forceinline__ v16h load_cvt_a(const float* __restrict__ rowk, int hi) {
    const v4f* p0 = (const v4f*)(rowk + hi * 16);
    const v4f* p1 = (const v4f*)(rowk + 8 + hi * 16);
    v4f f0 = __builtin_nontemporal_load(p0);
    v4f f1 = __builtin_nontemporal_load(p0 + 1);
    v4f f2 = __builtin_nontemporal_load(p1);
    v4f f3 = __builtin_nontemporal_load(p1 + 1);
    v16h a;
    a[0] = (_Float16)f0[0];  a[1] = (_Float16)f0[1];
    a[2] = (_Float16)f0[2];  a[3] = (_Float16)f0[3];
    a[4] = (_Float16)f1[0];  a[5] = (_Float16)f1[1];
    a[6] = (_Float16)f1[2];  a[7] = (_Float16)f1[3];
    a[8] = (_Float16)f2[0];  a[9] = (_Float16)f2[1];
    a[10] = (_Float16)f2[2]; a[11] = (_Float16)f2[3];
    a[12] = (_Float16)f3[0]; a[13] = (_Float16)f3[1];
    a[14] = (_Float16)f3[2]; a[15] = (_Float16)f3[3];
    return a;
}

// ---------------------------------------------------------------------------
// Kernel 0: W (row-major fp32 [NIN][NH]) -> Wcm (col-major f16, Wcm[h*NIN+d]);
// also zero the community-sum accumulator (needed fresh on every replay).
// ---------------------------------------------------------------------------
__global__ __launch_bounds__(256) void prep_kernel(const float* __restrict__ W,
                                                   _Float16* __restrict__ Wcm,
                                                   float* __restrict__ sums) {
    int idx = blockIdx.x * blockDim.x + threadIdx.x;
    if (idx < NIN * NH) {
        int d = idx >> 7;          // row of W
        int h = idx & 127;         // col of W
        Wcm[h * NIN + d] = (_Float16)W[idx];
    }
    if (idx < NC * NH) sums[idx] = 0.0f;
}

// ---------------------------------------------------------------------------
// Kernel 1: fts = seq @ W  (WMMA f16, f32 accumulate), output col-major f16:
// fts_cm[h*N + n]. Block = 4 waves, wave = 16 rows x 128 cols. grid.y = seq id.
// ---------------------------------------------------------------------------
__global__ __launch_bounds__(128) void fts_kernel(const float* __restrict__ seq1,
                                                  const float* __restrict__ seq2,
                                                  const _Float16* __restrict__ Wcm,
                                                  _Float16* __restrict__ f1,
                                                  _Float16* __restrict__ f2) {
    const float* seq = (blockIdx.y == 0) ? seq1 : seq2;
    _Float16* fcm    = (blockIdx.y == 0) ? f1 : f2;

    int wave = threadIdx.x >> 5;
    int lane = threadIdx.x & 31;
    int r = lane & 15, hi = lane >> 4;
    int row0 = blockIdx.x * 64 + wave * 16;

    v8f acc[8] = {};
    const float* arow = seq + (size_t)(row0 + r) * NIN;

#pragma unroll
    for (int k0 = 0; k0 < NIN; k0 += 32) {
        v16h a = load_cvt_a(arow + k0, hi);
#pragma unroll
        for (int ct = 0; ct < 8; ++ct) {
            int col = ct * 16 + r;
            v16h b = *(const v16h*)(Wcm + (size_t)col * NIN + k0 + hi * 16);
            acc[ct] = wmma_f16(a, b, acc[ct]);
        }
    }

    // C element j: row = row0 + hi*8 + j, col = ct*16 + r.
    // j=0..7 are consecutive rows -> pack to one 16B col-major store.
#pragma unroll
    for (int ct = 0; ct < 8; ++ct) {
        int col = ct * 16 + r;
        v8h pk;
#pragma unroll
        for (int j = 0; j < 8; ++j) pk[j] = (_Float16)acc[ct][j];
        *(v8h*)(fcm + (size_t)col * N_NODES + row0 + hi * 8) = pk;
    }
}

// ---------------------------------------------------------------------------
// Kernel 2: agg = adj @ fts + b, ReLU, for BOTH sequences sharing one adj read.
// Block = 4 waves sharing 16 adj rows (adj HBM traffic = 256MB total, once,
// non-temporal). Wave = 16 rows x 32 cols x 2 seqs = 4 WMMA accum tiles.
// 512 blocks -> 2048 waves.
// ---------------------------------------------------------------------------
__global__ __launch_bounds__(128) void agg_kernel(const float* __restrict__ adj,
                                                  const _Float16* __restrict__ f1,
                                                  const _Float16* __restrict__ f2,
                                                  const float* __restrict__ bias,
                                                  float* __restrict__ h1,
                                                  float* __restrict__ h2) {
    int wave = threadIdx.x >> 5;
    int lane = threadIdx.x & 31;
    int r = lane & 15, hi = lane >> 4;
    int row0 = blockIdx.x * 16;     // all 4 waves share these 16 rows (WGP$ reuse)
    int c0 = wave * 32;             // 2 col tiles per wave

    v8f acc[2][2] = {};             // [seq][col tile]
    const float* arow = adj + (size_t)(row0 + r) * N_NODES;

#pragma unroll 2
    for (int k0 = 0; k0 < N_NODES; k0 += 32) {
        __builtin_prefetch(arow + k0 + 1024, 0, 0);   // global_prefetch: adj stream
        v16h a = load_cvt_a(arow + k0, hi);
#pragma unroll
        for (int ct = 0; ct < 2; ++ct) {
            size_t boff = (size_t)(c0 + ct * 16 + r) * N_NODES + k0 + hi * 16;
            v16h b1 = *(const v16h*)(f1 + boff);
            acc[0][ct] = wmma_f16(a, b1, acc[0][ct]);
            v16h b2 = *(const v16h*)(f2 + boff);
            acc[1][ct] = wmma_f16(a, b2, acc[1][ct]);
        }
    }

#pragma unroll
    for (int s = 0; s < 2; ++s) {
        float* h = s ? h2 : h1;
#pragma unroll
        for (int ct = 0; ct < 2; ++ct) {
            int col = c0 + ct * 16 + r;
            float bv = bias[col];
#pragma unroll
            for (int j = 0; j < 8; ++j) {
                int row = row0 + hi * 8 + j;
                float v = acc[s][ct][j] + bv;
                h[(size_t)row * NH + col] = v > 0.0f ? v : 0.0f;
            }
        }
    }
}

// ---------------------------------------------------------------------------
// Kernel 3a: partial community sums of h1 (labels staged via LDS, atomic f32).
// grid (NC, 8 chunks), 128 threads (one per feature h).
// ---------------------------------------------------------------------------
__global__ __launch_bounds__(128) void comm_sum_kernel(const int* __restrict__ lab,
                                                       const float* __restrict__ h1,
                                                       float* __restrict__ sums) {
    __shared__ int nodes[128];
    int c = blockIdx.x, chunk = blockIdx.y, h = threadIdx.x;
    nodes[h] = lab[c * CS + chunk * 128 + h];
    __syncthreads();
    float s = 0.0f;
#pragma unroll 4
    for (int i = 0; i < 128; ++i) s += h1[(size_t)nodes[i] * NH + h];
    atomicAdd(&sums[c * NH + h], s);
}

// ---------------------------------------------------------------------------
// Kernel 3b: c = sigmoid(mean); cw[c][h] = sum_k Wb[h][k] * c[c][k].
// ---------------------------------------------------------------------------
__global__ __launch_bounds__(128) void cw_kernel(const float* __restrict__ sums,
                                                 const float* __restrict__ Wb,
                                                 float* __restrict__ cw) {
    __shared__ float cvec[NH];
    int c = blockIdx.x, h = threadIdx.x;
    float m = sums[c * NH + h] * (1.0f / (float)CS);
    cvec[h] = 1.0f / (1.0f + expf(-m));
    __syncthreads();
    float a = 0.0f;
#pragma unroll 4
    for (int k = 0; k < NH; ++k) a += Wb[h * NH + k] * cvec[k];
    cw[c * NH + h] = a;
}

// ---------------------------------------------------------------------------
// Kernel 4: per-node bilinear score + scatter. One wave (32 lanes) per node:
// 128-dot via v4f + __shfl_xor wave reduction. Every out element written.
// ---------------------------------------------------------------------------
__global__ __launch_bounds__(256) void score_kernel(const int* __restrict__ lab,
                                                    const float* __restrict__ h1,
                                                    const float* __restrict__ h2,
                                                    const float* __restrict__ cw,
                                                    const float* __restrict__ bbp,
                                                    float* __restrict__ out) {
    int wave = threadIdx.x >> 5;
    int lane = threadIdx.x & 31;
    int j = blockIdx.x * 8 + wave;       // flattened (community, slot) index
    int node = lab[j];
    int c = j >> 10;                     // j / CS

    const v4f* hp1 = (const v4f*)(h1 + (size_t)node * NH);
    const v4f* hp2 = (const v4f*)(h2 + (size_t)node * NH);
    const v4f* cwp = (const v4f*)(cw + (size_t)c * NH);
    v4f a1 = hp1[lane], a2 = hp2[lane], w = cwp[lane];
    float d1 = a1[0] * w[0] + a1[1] * w[1] + a1[2] * w[2] + a1[3] * w[3];
    float d2 = a2[0] * w[0] + a2[1] * w[1] + a2[2] * w[2] + a2[3] * w[3];
#pragma unroll
    for (int m = 16; m > 0; m >>= 1) {
        d1 += __shfl_xor(d1, m, 32);
        d2 += __shfl_xor(d2, m, 32);
    }
    if (lane == 0) {
        float bb = bbp[0];
        out[node] = d1 + bb;
        out[N_NODES + node] = d2 + bb;
    }
}

// ---------------------------------------------------------------------------
extern "C" void kernel_launch(void* const* d_in, const int* in_sizes, int n_in,
                              void* d_out, int out_size, void* d_ws, size_t ws_size,
                              hipStream_t stream) {
    const float* seq1 = (const float*)d_in[0];
    const float* seq2 = (const float*)d_in[1];
    const float* adj  = (const float*)d_in[2];
    const int*   lab  = (const int*)  d_in[3];
    const float* W    = (const float*)d_in[4];
    const float* b    = (const float*)d_in[5];
    const float* Wb   = (const float*)d_in[6];
    const float* bb   = (const float*)d_in[7];
    float* out = (float*)d_out;

    // Workspace layout (~12.04 MB total)
    char* ws = (char*)d_ws;
    _Float16* Wcm = (_Float16*)ws;                                 // 32 KB
    _Float16* f1  = (_Float16*)(ws + (32 << 10));                  // 2 MB
    _Float16* f2  = f1 + (size_t)N_NODES * NH;                     // 2 MB
    float* h1   = (float*)(ws + (32 << 10) +
                           2 * (size_t)N_NODES * NH * sizeof(_Float16)); // 4 MB
    float* h2   = h1 + (size_t)N_NODES * NH;                       // 4 MB
    float* sums = h2 + (size_t)N_NODES * NH;                       // 4 KB
    float* cw   = sums + NC * NH;                                  // 4 KB

    prep_kernel<<<64, 256, 0, stream>>>(W, Wcm, sums);
    fts_kernel<<<dim3(N_NODES / 64, 2), 128, 0, stream>>>(seq1, seq2, Wcm, f1, f2);
    agg_kernel<<<N_NODES / 16, 128, 0, stream>>>(adj, f1, f2, b, h1, h2);
    comm_sum_kernel<<<dim3(NC, 8), 128, 0, stream>>>(lab, h1, sums);
    cw_kernel<<<NC, 128, 0, stream>>>(sums, Wb, cw);
    score_kernel<<<N_NODES / 8, 256, 0, stream>>>(lab, h1, h2, cw, bb, out);
}
